// MultiHeadLatentAttention_7980049236626
// MI455X (gfx1250) — matrix operand order repaired and checked by reading, not compiled, not run on hardware
//
#include <hip/hip_runtime.h>
#include <stdint.h>

typedef __attribute__((ext_vector_type(16))) __bf16 v16bf;
typedef __attribute__((ext_vector_type(8)))  __bf16 v8bf;
typedef __attribute__((ext_vector_type(8)))  float  v8f;
typedef unsigned short u16;

#define BB   2
#define SS   2048
#define NH   16
#define HD   128
#define DM   2048
#define DL   512
#define DQK  256
#define MR   (BB*SS)   // 4096 token rows

// ---------- helpers ----------

__device__ __forceinline__ u16 f2bf(float f) {
  uint32_t u = __float_as_uint(f);
  return (u16)((u + 0x7FFFu + ((u >> 16) & 1u)) >> 16);   // round-to-nearest-even
}

// Build a 16-element bf16 fragment from two contiguous 8-element (16B) loads.
__device__ __forceinline__ v16bf make_frag(const u16* p, int gap) {
  v8bf lo = *(const v8bf*)p;
  v8bf hh = *(const v8bf*)(p + gap);
  v16bf r;
#pragma unroll
  for (int i = 0; i < 8; ++i) { r[i] = lo[i]; r[i + 8] = hh[i]; }
  return r;
}

// A fragment (16x32, row-major A): lane m=lane&15, hi=lane>>4.
// elem j<8 -> K = k0 + j + 8*hi ; elem j>=8 -> K = k0 + j + 8 + 8*hi.
__device__ __forceinline__ v16bf load_a_frag(const u16* A, int lda, int row0, int k0) {
  int lane = threadIdx.x & 31;
  const u16* p = A + (size_t)(row0 + (lane & 15)) * lda + k0 + 8 * (lane >> 4);
  return make_frag(p, 16);
}

__device__ __forceinline__ v8f wmma_bf16(v16bf a, v16bf b, v8f c) {
  return __builtin_amdgcn_wmma_f32_16x16x32_bf16(false, a, false, b, (short)0, c,
                                                 false, false);
}

// ---------- elementwise prep kernels ----------

__global__ void cvt_f32_bf16(const float* __restrict__ in, u16* __restrict__ out, int n) {
  int i = blockIdx.x * blockDim.x + threadIdx.x;
  if (i < n) out[i] = f2bf(in[i]);
}

// Tiled transpose+convert: Wt[n*K + k] = bf16(W[k*N + n]), coalesced both ways.
__global__ void __launch_bounds__(256)
transpose_f32_bf16(const float* __restrict__ W, u16* __restrict__ Wt, int K, int N) {
  __shared__ u16 tile[32][33];                 // padded: no bank conflicts
  int kt = blockIdx.x * 32, nt = blockIdx.y * 32;
  int tx = threadIdx.x & 31, ty = threadIdx.x >> 5;   // 32 x 8
#pragma unroll
  for (int r = 0; r < 32; r += 8)
    tile[r + ty][tx] = f2bf(W[(size_t)(kt + r + ty) * N + nt + tx]);
  __syncthreads();
#pragma unroll
  for (int r = 0; r < 32; r += 8)
    Wt[(size_t)(nt + r + ty) * K + kt + tx] = tile[tx][r + ty];
}

// ---------- bf16 WMMA GEMM : C[M,N] = A[M,K] * Bt[N,K]^T ----------
// One wave owns a 32x64 C tile (8 accumulators). Leading dims are template
// constants so every fragment address is (persistent pointer + immediate):
// two pointer increments per K-step, no VALU address rebuild, no WAR-hazard
// NOPs against WMMA sources. Software pipelined with a peeled tail.

template <int LDA, int LDBT>
__global__ void __launch_bounds__(128)
gemm_bf16_wmma(const u16* __restrict__ A, const u16* __restrict__ Bt,
               float* __restrict__ Cf, u16* __restrict__ Cb, int ldc,
               int M, int N, int K) {
  int wg  = blockIdx.x * 4 + (threadIdx.x >> 5);
  int ntn = N >> 6;                       // 64-wide N tiles
  int tm = wg / ntn, tn = wg % ntn;
  if (tm * 32 >= M) return;
  int lane = threadIdx.x & 31, nl = lane & 15, hi = lane >> 4;
  int m0 = tm * 32, n0 = tn * 64;

  const u16* pa = A  + (size_t)(m0 + nl) * LDA  + 8  * hi;   // A row m0+nl
  const u16* pb = Bt + (size_t)(n0 + nl) * LDBT + 8  * hi;   // Bt row n0+nl

  v8f acc[8] = {};
  v16bf ca0 = make_frag(pa,            16);
  v16bf ca1 = make_frag(pa + 16 * LDA, 16);
  v16bf cb0 = make_frag(pb,             16);
  v16bf cb1 = make_frag(pb + 16 * LDBT, 16);
  v16bf cb2 = make_frag(pb + 32 * LDBT, 16);
  v16bf cb3 = make_frag(pb + 48 * LDBT, 16);

  for (int it = 0; it < K / 32 - 1; ++it) {
    pa += 32; pb += 32;
    v16bf na0 = make_frag(pa,            16);
    v16bf na1 = make_frag(pa + 16 * LDA, 16);
    v16bf nb0 = make_frag(pb,             16);
    v16bf nb1 = make_frag(pb + 16 * LDBT, 16);
    v16bf nb2 = make_frag(pb + 32 * LDBT, 16);
    v16bf nb3 = make_frag(pb + 48 * LDBT, 16);

    acc[0] = wmma_bf16(ca0, cb0, acc[0]);
    acc[1] = wmma_bf16(ca0, cb1, acc[1]);
    acc[2] = wmma_bf16(ca0, cb2, acc[2]);
    acc[3] = wmma_bf16(ca0, cb3, acc[3]);
    acc[4] = wmma_bf16(ca1, cb0, acc[4]);
    acc[5] = wmma_bf16(ca1, cb1, acc[5]);
    acc[6] = wmma_bf16(ca1, cb2, acc[6]);
    acc[7] = wmma_bf16(ca1, cb3, acc[7]);

    ca0 = na0; ca1 = na1;
    cb0 = nb0; cb1 = nb1; cb2 = nb2; cb3 = nb3;
  }
  acc[0] = wmma_bf16(ca0, cb0, acc[0]);
  acc[1] = wmma_bf16(ca0, cb1, acc[1]);
  acc[2] = wmma_bf16(ca0, cb2, acc[2]);
  acc[3] = wmma_bf16(ca0, cb3, acc[3]);
  acc[4] = wmma_bf16(ca1, cb0, acc[4]);
  acc[5] = wmma_bf16(ca1, cb1, acc[5]);
  acc[6] = wmma_bf16(ca1, cb2, acc[6]);
  acc[7] = wmma_bf16(ca1, cb3, acc[7]);

  // C layout: VGPR r holds (M = r + 8*hi, N = lane&15)
#pragma unroll
  for (int half = 0; half < 2; ++half)
#pragma unroll
    for (int t = 0; t < 4; ++t)
#pragma unroll
      for (int r = 0; r < 8; ++r) {
        size_t row = (size_t)m0 + half * 16 + r + 8 * hi;
        size_t col = (size_t)n0 + t * 16 + nl;
        float v = acc[half * 4 + t][r];
        if (Cb) Cb[row * ldc + col] = f2bf(v);
        else    Cf[row * ldc + col] = v;
      }
}

// ---------- Q/K assembly + RoPE, V transpose ----------
// reference: q = concat(q_no_rope[1024], q_rope[1024]).reshape(16,128);
// head h <- concat[h*128 : h*128+128]; RoPE on dims 64..127 of each head.

__global__ void assemble_qkv(const float* __restrict__ qnr, const float* __restrict__ knr,
                             const float* __restrict__ qr,  const float* __restrict__ kr,
                             const float* __restrict__ vf,
                             u16* __restrict__ Qb, u16* __restrict__ Kb,
                             u16* __restrict__ Vt) {
  int idx = blockIdx.x * blockDim.x + threadIdx.x;   // (b,s,h,d)
  if (idx >= MR * NH * HD) return;
  int d   = idx & 127;
  int h   = (idx >> 7) & 15;
  int row = idx >> 11;                 // b*SS + s
  int s   = row & (SS - 1);
  int b   = row >> 11;
  const size_t r1024 = (size_t)row * 1024;
  auto fetch = [&](const float* nr, const float* rp, int j) -> float {
    return (j < 1024) ? nr[r1024 + j] : rp[r1024 + j - 1024];
  };
  float qv, kv;
  if (d < 64) {
    int j = h * 128 + d;
    qv = fetch(qnr, qr, j);
    kv = fetch(knr, kr, j);
  } else {
    int dd = d - 64;
    int i  = dd & 31;                  // frequency index, half = 32
    // inv_freq[i] = 10000^(-i/32)
    float ang = (float)s * __expf((float)i * (-9.210340371976184f / 32.0f));
    float cs = __cosf(ang), sn = __sinf(ang);
    int j1 = h * 128 + 64 + i, j2 = h * 128 + 96 + i;
    float q1 = fetch(qnr, qr, j1), q2 = fetch(qnr, qr, j2);
    float k1 = fetch(knr, kr, j1), k2 = fetch(knr, kr, j2);
    if (dd < 32) { qv = q1 * cs - q2 * sn; kv = k1 * cs - k2 * sn; }
    else         { qv = q1 * sn + q2 * cs; kv = k1 * sn + k2 * cs; }
  }
  size_t bh   = (size_t)(b * NH + h);
  size_t bhsd = (bh * SS + s) * HD + d;
  Qb[bhsd] = f2bf(qv);
  Kb[bhsd] = f2bf(kv);
  Vt[(bh * HD + d) * SS + s] = f2bf(vf[(size_t)row * DM + h * 128 + d]);  // V^T
}

// ---------- flash attention: 1 wave = (b,h) x 16 query rows ----------

__global__ void __launch_bounds__(128)
flash_attn_wmma(const u16* __restrict__ Qb, const u16* __restrict__ Kb,
                const u16* __restrict__ Vt, u16* __restrict__ ctxb) {
  __shared__ u16 plds[4][16 * 32];       // per-wave P staging (C-layout -> A-layout)
  int wid  = threadIdx.x >> 5;
  int lane = threadIdx.x & 31, nl = lane & 15, hi = lane >> 4;
  int wg = blockIdx.x * 4 + wid;
  int qt = wg & 127;                     // SS/16 = 128 q tiles per head
  int bh = wg >> 7;
  int b = bh >> 4, h = bh & 15;
  const u16* q  = Qb + (size_t)bh * SS * HD;

  v16bf qa[4];
#pragma unroll
  for (int i = 0; i < 4; ++i) qa[i] = load_a_frag(q, HD, qt * 16, i * 32);

  // persistent per-lane pointers; all fragment offsets are immediates
  const u16* kptr = Kb + (size_t)bh * SS * HD + (size_t)nl * HD + 8 * hi;
  const u16* vptr = Vt + (size_t)bh * HD * SS + (size_t)nl * SS + 8 * hi;

  float mi[8], li[8];
#pragma unroll
  for (int r = 0; r < 8; ++r) { mi[r] = -1e30f; li[r] = 0.f; }
  v8f o[8] = {};
  const float scale = 0.08838834764831845f;   // 1/sqrt(128)

  for (int kt = 0; kt < SS / 32; ++kt) {
    // ---- batched K-fragment loads, then 8 back-to-back score WMMAs ----
    // B fragment for scores: key row = kt*32 + 16*g + nl, dim offset 32*i
    v16bf kb[8];
#pragma unroll
    for (int i = 0; i < 4; ++i) {
      kb[2 * i]     = make_frag(kptr + 32 * i,            16);
      kb[2 * i + 1] = make_frag(kptr + 16 * HD + 32 * i,  16);
    }
    v8f s0 = {}, s1 = {};
#pragma unroll
    for (int i = 0; i < 4; ++i) {
      s0 = wmma_bf16(qa[i], kb[2 * i],     s0);
      s1 = wmma_bf16(qa[i], kb[2 * i + 1], s1);
    }
    // ---- issue V-fragment loads now: they fly behind the softmax VALU work ----
    v16bf bv[8];
#pragma unroll
    for (int nb = 0; nb < 8; ++nb)
      bv[nb] = make_frag(vptr + (size_t)nb * 16 * SS, 16);

    // ---- online softmax; row M = r + 8*hi lives in a 16-lane group ----
    float alpha[8];
#pragma unroll
    for (int r = 0; r < 8; ++r) {
      float x0 = s0[r] * scale, x1 = s1[r] * scale;
      float mx = fmaxf(x0, x1);
#pragma unroll
      for (int off = 8; off; off >>= 1) mx = fmaxf(mx, __shfl_xor(mx, off, 32));
      float mn = fmaxf(mi[r], mx);
      float a  = __expf(mi[r] - mn);
      float p0 = __expf(x0 - mn), p1 = __expf(x1 - mn);
      float rs = p0 + p1;
#pragma unroll
      for (int off = 8; off; off >>= 1) rs += __shfl_xor(rs, off, 32);
      li[r] = li[r] * a + rs;
      mi[r] = mn;
      alpha[r] = a;
      plds[wid][(r + 8 * hi) * 32 + nl]      = f2bf(p0);
      plds[wid][(r + 8 * hi) * 32 + 16 + nl] = f2bf(p1);
    }
    asm volatile("s_wait_dscnt 0" ::: "memory");   // wave-local LDS RAW fence
    v16bf pa = make_frag(&plds[wid][nl * 32 + 8 * hi], 16);   // A-layout reload

    // ---- o = o*alpha + P @ V ----
#pragma unroll
    for (int nb = 0; nb < 8; ++nb) {
#pragma unroll
      for (int r = 0; r < 8; ++r) o[nb][r] *= alpha[r];
      o[nb] = wmma_bf16(pa, bv[nb], o[nb]);
    }
    kptr += 32 * HD;     // next 32 keys
    vptr += 32;
  }
  float inv[8];
#pragma unroll
  for (int r = 0; r < 8; ++r) inv[r] = 1.0f / li[r];
#pragma unroll
  for (int nb = 0; nb < 8; ++nb)
#pragma unroll
    for (int r = 0; r < 8; ++r) {
      int srow = qt * 16 + r + 8 * hi;
      ctxb[(size_t)(b * SS + srow) * DM + h * 128 + nb * 16 + nl] =
          f2bf(o[nb][r] * inv[r]);
    }
}

// ---------- host orchestration ----------

extern "C" void kernel_launch(void* const* d_in, const int* in_sizes, int n_in,
                              void* d_out, int out_size, void* d_ws, size_t ws_size,
                              hipStream_t stream) {
  (void)in_sizes; (void)n_in; (void)out_size; (void)ws_size;
  const float* x      = (const float*)d_in[0];
  const float* Wcomp  = (const float*)d_in[1];
  const float* Wqdec  = (const float*)d_in[2];
  const float* Wkdec  = (const float*)d_in[3];
  const float* Wvdec  = (const float*)d_in[4];
  const float* Wropeq = (const float*)d_in[5];
  const float* Wropek = (const float*)d_in[6];
  const float* Wout   = (const float*)d_in[7];
  float* out = (float*)d_out;

  uint8_t* base = (uint8_t*)d_ws;
  size_t off = 0;
  auto alloc = [&](size_t bytes) -> uint8_t* {
    uint8_t* r = base + off;
    off += (bytes + 255) & ~(size_t)255;
    return r;
  };
  u16*   xb    = (u16*)alloc((size_t)MR * DM * 2);
  u16*   Wct   = (u16*)alloc((size_t)DL * DM * 2);
  u16*   Wqt   = (u16*)alloc((size_t)1024 * DQK * 2);
  u16*   Wkt   = (u16*)alloc((size_t)1024 * DQK * 2);
  u16*   Wvt   = (u16*)alloc((size_t)2048 * DQK * 2);
  u16*   Wrqt  = (u16*)alloc((size_t)1024 * DM * 2);
  u16*   Wrkt  = (u16*)alloc((size_t)1024 * DM * 2);
  u16*   Wot   = (u16*)alloc((size_t)DM * DM * 2);
  u16*   compb = (u16*)alloc((size_t)MR * DL * 2);
  float* qnr   = (float*)alloc((size_t)MR * 1024 * 4);
  float* knr   = (float*)alloc((size_t)MR * 1024 * 4);
  float* qr    = (float*)alloc((size_t)MR * 1024 * 4);
  float* kr    = (float*)alloc((size_t)MR * 1024 * 4);
  float* vf    = (float*)alloc((size_t)MR * DM * 4);
  u16*   Qb    = (u16*)alloc((size_t)MR * DM * 2);
  u16*   Kb    = (u16*)alloc((size_t)MR * DM * 2);
  u16*   Vt    = (u16*)alloc((size_t)MR * DM * 2);
  u16*   ctxb  = (u16*)alloc((size_t)MR * DM * 2);

  { int n = MR * DM;
    cvt_f32_bf16<<<(n + 255) / 256, 256, 0, stream>>>(x, xb, n); }

  auto tr = [&](const float* W, u16* Wt, int K, int N) {
    dim3 g(K / 32, N / 32);
    transpose_f32_bf16<<<g, 256, 0, stream>>>(W, Wt, K, N);
  };
  tr(Wcomp,  Wct,  DM,  DL);
  tr(Wqdec,  Wqt,  DQK, 1024);
  tr(Wkdec,  Wkt,  DQK, 1024);
  tr(Wvdec,  Wvt,  DQK, 2048);
  tr(Wropeq, Wrqt, DM,  1024);
  tr(Wropek, Wrkt, DM,  1024);
  tr(Wout,   Wot,  DM,  DM);

  // big GEMMs: lda = 2048, ldbt = 2048
  auto gemm_big = [&](const u16* A, const u16* Bt, float* Cf, u16* Cb, int ldc,
                      int M, int N, int K) {
    int waves = (M / 32) * (N / 64);
    gemm_bf16_wmma<DM, DM><<<waves / 4, 128, 0, stream>>>(A, Bt, Cf, Cb, ldc, M, N, K);
  };
  // decode GEMMs: lda = 512, ldbt = 256
  auto gemm_dec = [&](const u16* A, const u16* Bt, float* Cf, u16* Cb, int ldc,
                      int M, int N, int K) {
    int waves = (M / 32) * (N / 64);
    gemm_bf16_wmma<DL, DQK><<<waves / 4, 128, 0, stream>>>(A, Bt, Cf, Cb, ldc, M, N, K);
  };

  gemm_big(xb, Wct, nullptr, compb, DL, MR, DL, DM);          // compressed (bf16)
  gemm_dec(compb,       Wqt, qnr, nullptr, 1024, MR, 1024, DQK); // q_no_rope
  gemm_dec(compb,       Wkt, knr, nullptr, 1024, MR, 1024, DQK); // k_no_rope
  gemm_dec(compb + DQK, Wvt, vf,  nullptr, DM,   MR, DM,   DQK); // v
  gemm_big(xb, Wrqt, qr, nullptr, 1024, MR, 1024, DM);           // q_rope
  gemm_big(xb, Wrkt, kr, nullptr, 1024, MR, 1024, DM);           // k_rope

  { int n = MR * NH * HD;
    assemble_qkv<<<(n + 255) / 256, 256, 0, stream>>>(qnr, knr, qr, kr, vf, Qb, Kb, Vt); }

  { int waves = BB * NH * (SS / 16);   // 4096 waves
    flash_attn_wmma<<<waves / 4, 128, 0, stream>>>(Qb, Kb, Vt, ctxb); }

  gemm_big(ctxb, Wot, out, nullptr, DM, MR, DM, DM);             // output proj (f32)
}
